// AttentionMemoryEntry_34505767256886
// MI455X (gfx1250) — matrix operand order repaired and checked by reading, not compiled, not run on hardware
//
#include <hip/hip_runtime.h>
#include <hip/hip_bf16.h>

// ---------------- problem constants ----------------
#define Bc    4
#define Lc    256
#define NMEMc 64
#define LMEMc 256
#define Dc    1024
#define Hc    16
#define DFFc  4096
#define DKc   64
#define Tc    (Bc*Lc)          // 1024 tokens
#define MKc   (NMEMc*LMEMc)    // 16384 memory rows

typedef __attribute__((ext_vector_type(16))) _Float16 v16h;
typedef __attribute__((ext_vector_type(8)))  _Float16 v8h;
typedef __attribute__((ext_vector_type(8)))  float    v8f;
typedef __attribute__((ext_vector_type(4)))  int      i4;

// LDS strides in halves (pitch 80B: 16B aligned, spreads banks)
#define LDA 40
#define LDW 40

// ---------------- CDNA5 async global->LDS (guarded) ----------------
#if __has_builtin(__builtin_amdgcn_global_load_async_to_lds_b128)
#define HAVE_ASYNC_LDS 1
#else
#define HAVE_ASYNC_LDS 0
#endif

__device__ __forceinline__ void stage_b128(const void* g, void* l) {
#if HAVE_ASYNC_LDS
    __builtin_amdgcn_global_load_async_to_lds_b128(
        (__attribute__((address_space(1))) i4*)g,
        (__attribute__((address_space(3))) i4*)l, 0, 0);
#else
    *(v8h*)l = *(const v8h*)g;
#endif
}

template<int N>
__device__ __forceinline__ void stage_wait_n() {
#if HAVE_ASYNC_LDS
#if __has_builtin(__builtin_amdgcn_s_wait_asynccnt)
    __builtin_amdgcn_s_wait_asynccnt(N);
#else
    asm volatile("s_wait_asynccnt %0" :: "i"(N) : "memory");
#endif
#endif
}

// ============================================================
// WMMA GEMM:  out = epi(A[M,K] @ WT[N,K]^T + bias [+res])
//   AF16: A is f16 (async b128 staging); else f32 (convert while staging)
//   WT is f16, pre-transposed [N][K].
// Block: 256 threads (8 waves, 4x2). Block tile 128x128, wave tile 32x64.
// Double-buffered async LDS staging (prefetch next k-tile while computing).
// Requires M%128==0, N%128==0, K%32==0 (true for all shapes here).
// ============================================================
template<bool AF16, bool RELU, bool HASRES, bool OUTF, bool OUTH>
__global__ __launch_bounds__(256)
void gemm_f16wmma_kernel(const void* __restrict__ Aop,
                         const _Float16* __restrict__ WT,
                         const float* __restrict__ bias,
                         const float* __restrict__ res,
                         float* __restrict__ outf,
                         _Float16* __restrict__ outh,
                         int M, int N, int K)
{
    __shared__ __align__(16) _Float16 As[2][128 * LDA];   // [row][k]
    __shared__ __align__(16) _Float16 Wt[2][128 * LDW];   // [col][k]

    const int tid  = threadIdx.x;
    const int wave = tid >> 5;
    const int lane = tid & 31;

    const int rowBase = blockIdx.y * 128;
    const int colBase = blockIdx.x * 128;
    const int wr = (wave >> 1) * 32;     // wave row offset in block tile
    const int wc = (wave & 1) * 64;      // wave col offset in block tile

    const int halfSel = lane >> 4;
    const int mrow    = lane & 15;
    const int ncol    = lane & 15;
    const int kbA     = halfSel * 8;     // A frag: K {kbA..+7, 16+kbA..+7}
    const int kbB     = halfSel * 8;     // B frag: K {kbB..+7, 16+kbB..+7}

    // per-thread staging coordinates (128x32 halves each for A and W)
    const int ar = tid >> 1, ac = (tid & 1) * 16;
    const int wn = tid >> 1, wk2 = (tid & 1) * 16;

    auto stage = [&](int buf, int k0) {
        if constexpr (AF16) {
            const _Float16* src = (const _Float16*)Aop + (size_t)(rowBase + ar) * K + k0 + ac;
            _Float16* dst = &As[buf][ar * LDA + ac];
            stage_b128(src, dst);
            stage_b128(src + 8, dst + 8);
        } else {
            const float* src = (const float*)Aop + (size_t)(rowBase + ar) * K + k0 + ac;
            _Float16* dst = &As[buf][ar * LDA + ac];
            #pragma unroll
            for (int i = 0; i < 16; ++i) dst[i] = (_Float16)src[i];
        }
        const _Float16* wsrc = WT + (size_t)(colBase + wn) * K + k0 + wk2;
        _Float16* wdst = &Wt[buf][wn * LDW + wk2];
        stage_b128(wsrc, wdst);
        stage_b128(wsrc + 8, wdst + 8);
    };

    const v8f vz = {0.f,0.f,0.f,0.f,0.f,0.f,0.f,0.f};
    v8f acc[8] = {vz, vz, vz, vz, vz, vz, vz, vz};   // [sub(2)][nt(4)]

    int cur = 0;
    stage(0, 0);
    for (int k0 = 0; k0 < K; k0 += 32) {
        if (k0 + 32 < K) {                    // prefetch next tile into other buffer
            stage(cur ^ 1, k0 + 32);
            stage_wait_n<AF16 ? 4 : 2>();     // current tile's asyncs complete
        } else {
            stage_wait_n<0>();
        }
        __syncthreads();

        // ---- 2 A fragments (documented CDNA5 16-bit A layout) ----
        v16h af[2];
        #pragma unroll
        for (int s = 0; s < 2; ++s) {
            const _Float16* arow = &As[cur][(wr + s * 16 + mrow) * LDA];
            v8h alo = *(const v8h*)(arow + kbA);
            v8h ahi = *(const v8h*)(arow + 16 + kbA);
            af[s] = __builtin_shufflevector(alo, ahi,
                      0,1,2,3,4,5,6,7, 8,9,10,11,12,13,14,15);
        }
        // ---- 4 B fragments ----
        v16h bfr[4];
        #pragma unroll
        for (int nt = 0; nt < 4; ++nt) {
            const _Float16* wrow = &Wt[cur][(wc + nt * 16 + ncol) * LDW + kbB];
            v8h blo = *(const v8h*)(wrow);
            v8h bhi = *(const v8h*)(wrow + 16);
            bfr[nt] = __builtin_shufflevector(blo, bhi,
                        0,1,2,3,4,5,6,7, 8,9,10,11,12,13,14,15);
        }
        // ---- 8 independent WMMAs ----
        #pragma unroll
        for (int s = 0; s < 2; ++s)
            #pragma unroll
            for (int nt = 0; nt < 4; ++nt)
                acc[s * 4 + nt] = __builtin_amdgcn_wmma_f32_16x16x32_f16(
                    false, af[s], false, bfr[nt], (short)0, acc[s * 4 + nt], false, false);

        __syncthreads();
        cur ^= 1;
    }

    // ---- branch-free epilogue ----
    #pragma unroll
    for (int s = 0; s < 2; ++s) {
        #pragma unroll
        for (int nt = 0; nt < 4; ++nt) {
            const int col = colBase + wc + nt * 16 + ncol;
            const float bv = bias[col];
            #pragma unroll
            for (int r = 0; r < 8; ++r) {
                const int row = rowBase + wr + s * 16 + halfSel * 8 + r;
                float v = acc[s * 4 + nt][r] + bv;
                if constexpr (RELU)   v = fmaxf(v, 0.0f);
                if constexpr (HASRES) v += res[(size_t)row * N + col];
                if constexpr (OUTF)   outf[(size_t)row * N + col] = v;
                if constexpr (OUTH)   outh[(size_t)row * N + col] = (_Float16)v;
            }
        }
    }
}

// ============================================================
// One-time weight transpose + f32->f16: in[K][N] -> out[N][K]
// ============================================================
__global__ __launch_bounds__(256)
void wT_convert_kernel(const float* __restrict__ in, _Float16* __restrict__ out,
                       int K, int N)
{
    __shared__ _Float16 t[32][33];
    const int tx = threadIdx.x & 31, ty = threadIdx.x >> 5;
    const int nb = blockIdx.x * 32, kb = blockIdx.y * 32;
    #pragma unroll
    for (int r = 0; r < 4; ++r)
        t[ty + 8*r][tx] = (_Float16)in[(size_t)(kb + ty + 8*r) * N + nb + tx];
    __syncthreads();
    #pragma unroll
    for (int r = 0; r < 4; ++r)
        out[(size_t)(nb + ty + 8*r) * K + kb + tx] = t[tx][ty + 8*r];
}

// ============================================================
// argmax routing (first-max tie rule like jnp.argmax)
// ============================================================
__global__ void route_argmax_kernel(const float* __restrict__ ma,
                                    int* __restrict__ samples)
{
    const int t = blockIdx.x * blockDim.x + threadIdx.x;
    if (t >= Tc) return;
    const float* p = ma + (size_t)t * NMEMc;
    float best = p[0]; int bi = 0;
    for (int j = 1; j < NMEMc; ++j) { float v = p[j]; if (v > best) { best = v; bi = j; } }
    samples[t] = bi;
}

// ============================================================
// LayerNorm: one row per block; writes f32 and f16 copies
// ============================================================
__global__ __launch_bounds__(256)
void layernorm_kernel(const float* __restrict__ in,
                      const float* __restrict__ g,
                      const float* __restrict__ be,
                      float* __restrict__ out, _Float16* __restrict__ outh)
{
    __shared__ float r1[256], r2[256];
    const int row = blockIdx.x, tid = threadIdx.x;
    const float* x = in + (size_t)row * Dc;
    float s = 0.f, s2 = 0.f;
    #pragma unroll
    for (int i = tid; i < Dc; i += 256) { float v = x[i]; s += v; s2 += v * v; }
    r1[tid] = s; r2[tid] = s2; __syncthreads();
    for (int off = 128; off > 0; off >>= 1) {
        if (tid < off) { r1[tid] += r1[tid + off]; r2[tid] += r2[tid + off]; }
        __syncthreads();
    }
    const float mean = r1[0] * (1.0f / Dc);
    const float var  = r2[0] * (1.0f / Dc) - mean * mean;
    const float rstd = rsqrtf(var + 1e-5f);
    #pragma unroll
    for (int i = tid; i < Dc; i += 256) {
        const float v = (x[i] - mean) * rstd * g[i] + be[i];
        out[(size_t)row * Dc + i]  = v;
        outh[(size_t)row * Dc + i] = (_Float16)v;
    }
}

// ============================================================
// y = x + sigmoid(l1-l0) * LN(st1; g1, be1); writes f32 + f16
// ============================================================
__global__ __launch_bounds__(256)
void ln_gate_kernel(const float* __restrict__ st1,
                    const float* __restrict__ g,
                    const float* __restrict__ be,
                    const float* __restrict__ x,
                    const float* __restrict__ gate_logits,
                    float* __restrict__ y, _Float16* __restrict__ yh)
{
    __shared__ float r1[256], r2[256];
    const int row = blockIdx.x, tid = threadIdx.x;
    const float* xr = st1 + (size_t)row * Dc;
    float s = 0.f, s2 = 0.f;
    #pragma unroll
    for (int i = tid; i < Dc; i += 256) { float v = xr[i]; s += v; s2 += v * v; }
    r1[tid] = s; r2[tid] = s2; __syncthreads();
    for (int off = 128; off > 0; off >>= 1) {
        if (tid < off) { r1[tid] += r1[tid + off]; r2[tid] += r2[tid + off]; }
        __syncthreads();
    }
    const float mean = r1[0] * (1.0f / Dc);
    const float var  = r2[0] * (1.0f / Dc) - mean * mean;
    const float rstd = rsqrtf(var + 1e-5f);
    const float l0 = gate_logits[(size_t)row * 2 + 0];
    const float l1 = gate_logits[(size_t)row * 2 + 1];
    const float gate = 1.0f / (1.0f + __expf(l0 - l1));
    #pragma unroll
    for (int i = tid; i < Dc; i += 256) {
        const float st2 = (xr[i] - mean) * rstd * g[i] + be[i];
        const float v = x[(size_t)row * Dc + i] + gate * st2;
        y[(size_t)row * Dc + i]  = v;
        yh[(size_t)row * Dc + i] = (_Float16)v;
    }
}

// ============================================================
// Per-token single-query attention vs routed memory entry.
// One block per token (256 threads). Writes f16 context.
// ============================================================
__global__ __launch_bounds__(256)
void routed_attn_kernel(const float* __restrict__ q,
                        const _Float16* __restrict__ kmem,
                        const _Float16* __restrict__ vmem,
                        const int* __restrict__ samples,
                        const unsigned char* __restrict__ maskmem,
                        _Float16* __restrict__ ctxh)
{
    __shared__ float qs[Dc];
    __shared__ float sc[Hc * LMEMc];
    __shared__ float red[256];

    const int t = blockIdx.x, tid = threadIdx.x;
    const int s = samples[t];
    const _Float16* kb = kmem + (size_t)s * LMEMc * Dc;
    const _Float16* vb = vmem + (size_t)s * LMEMc * Dc;
    const unsigned char* mrow = maskmem + (size_t)s * LMEMc;

    #pragma unroll
    for (int i = tid; i < Dc; i += 256) qs[i] = q[(size_t)t * Dc + i] * 0.125f;
    __syncthreads();

    #pragma unroll
    for (int p = tid; p < Hc * LMEMc; p += 256) {
        const int h = p >> 8, m = p & (LMEMc - 1);
        const _Float16* kr = kb + (size_t)m * Dc + h * DKc;
        const float* qh = qs + h * DKc;
        float a = 0.f;
        #pragma unroll 8
        for (int d = 0; d < DKc; ++d) a += qh[d] * (float)kr[d];
        sc[p] = mrow[m] ? a : -1e9f;
    }
    __syncthreads();

    for (int h = 0; h < Hc; ++h) {
        const float v = sc[h * LMEMc + tid];
        red[tid] = v; __syncthreads();
        for (int off = 128; off > 0; off >>= 1) {
            if (tid < off) red[tid] = fmaxf(red[tid], red[tid + off]);
            __syncthreads();
        }
        const float mx = red[0]; __syncthreads();
        const float e = __expf(v - mx);
        red[tid] = e; __syncthreads();
        for (int off = 128; off > 0; off >>= 1) {
            if (tid < off) red[tid] += red[tid + off];
            __syncthreads();
        }
        const float inv = 1.0f / red[0]; __syncthreads();
        sc[h * LMEMc + tid] = e * inv;
    }
    __syncthreads();

    {
        const int h = tid >> 6, d = tid & 63;
        const _Float16* vcol = vb + h * DKc + d;
        const float* a = sc + h * LMEMc;
        float accv = 0.f;
        #pragma unroll 8
        for (int m = 0; m < LMEMc; ++m) accv += a[m] * (float)vcol[(size_t)m * Dc];
        ctxh[(size_t)t * Dc + tid] = (_Float16)accv;
    }
}

// ============================================================
extern "C" void kernel_launch(void* const* d_in, const int* in_sizes, int n_in,
                              void* d_out, int out_size, void* d_ws, size_t ws_size,
                              hipStream_t stream) {
    (void)in_sizes; (void)n_in; (void)out_size; (void)ws_size;

    const float* dec  = (const float*)d_in[0];
    const float* gl   = (const float*)d_in[1];
    const float* ma   = (const float*)d_in[2];
    const float* enc  = (const float*)d_in[3];
    const float* tgt  = (const float*)d_in[4];
    const unsigned char* maskmem = (const unsigned char*)d_in[6];
    const float* wq = (const float*)d_in[7],  *bq = (const float*)d_in[8];
    const float* wk = (const float*)d_in[9],  *bk = (const float*)d_in[10];
    const float* wv = (const float*)d_in[11], *bv = (const float*)d_in[12];
    const float* wo = (const float*)d_in[13], *bo = (const float*)d_in[14];
    const float* w1a = (const float*)d_in[15], *b1a = (const float*)d_in[16];
    const float* w2a = (const float*)d_in[17], *b2a = (const float*)d_in[18];
    const float* w1b = (const float*)d_in[19], *b1b = (const float*)d_in[20];
    const float* w2b = (const float*)d_in[21], *b2b = (const float*)d_in[22];
    const float* g0 = (const float*)d_in[23], *be0 = (const float*)d_in[24];
    const float* g1 = (const float*)d_in[25], *be1 = (const float*)d_in[26];

    // -------- workspace layout --------
    char* p = (char*)d_ws;
    float* x   = (float*)p; p += (size_t)Tc * Dc * 4;
    float* qb  = (float*)p; p += (size_t)Tc * Dc * 4;
    float* st0 = (float*)p; p += (size_t)Tc * Dc * 4;
    float* st1 = (float*)p; p += (size_t)Tc * Dc * 4;
    float* yb  = (float*)p; p += (size_t)Tc * Dc * 4;
    _Float16* xh   = (_Float16*)p; p += (size_t)Tc * Dc * 2;
    _Float16* ctxh = (_Float16*)p; p += (size_t)Tc * Dc * 2;
    _Float16* st0h = (_Float16*)p; p += (size_t)Tc * Dc * 2;
    _Float16* yh   = (_Float16*)p; p += (size_t)Tc * Dc * 2;
    _Float16* hh   = (_Float16*)p; p += (size_t)Tc * DFFc * 2;
    _Float16* kf16 = (_Float16*)p; p += (size_t)MKc * Dc * 2;
    _Float16* vf16 = (_Float16*)p; p += (size_t)MKc * Dc * 2;
    _Float16* wqT  = (_Float16*)p; p += (size_t)Dc * Dc * 2;
    _Float16* wkT  = (_Float16*)p; p += (size_t)Dc * Dc * 2;
    _Float16* wvT  = (_Float16*)p; p += (size_t)Dc * Dc * 2;
    _Float16* woT  = (_Float16*)p; p += (size_t)Dc * Dc * 2;
    _Float16* w1aT = (_Float16*)p; p += (size_t)Dc * DFFc * 2;
    _Float16* w2aT = (_Float16*)p; p += (size_t)Dc * DFFc * 2;
    _Float16* w1bT = (_Float16*)p; p += (size_t)Dc * DFFc * 2;
    _Float16* w2bT = (_Float16*)p; p += (size_t)Dc * DFFc * 2;
    int* smp = (int*)p; p += (size_t)Tc * 4;

    const dim3 blk(256);

    // -------- one-time weight transpose+convert to f16 [N][K] --------
    wT_convert_kernel<<<dim3(Dc/32,  Dc/32),  blk, 0, stream>>>(wq,  wqT,  Dc,  Dc);
    wT_convert_kernel<<<dim3(Dc/32,  Dc/32),  blk, 0, stream>>>(wk,  wkT,  Dc,  Dc);
    wT_convert_kernel<<<dim3(Dc/32,  Dc/32),  blk, 0, stream>>>(wv,  wvT,  Dc,  Dc);
    wT_convert_kernel<<<dim3(Dc/32,  Dc/32),  blk, 0, stream>>>(wo,  woT,  Dc,  Dc);
    wT_convert_kernel<<<dim3(DFFc/32,Dc/32),  blk, 0, stream>>>(w1a, w1aT, Dc,  DFFc);
    wT_convert_kernel<<<dim3(Dc/32,  DFFc/32),blk, 0, stream>>>(w2a, w2aT, DFFc,Dc);
    wT_convert_kernel<<<dim3(DFFc/32,Dc/32),  blk, 0, stream>>>(w1b, w1bT, Dc,  DFFc);
    wT_convert_kernel<<<dim3(Dc/32,  DFFc/32),blk, 0, stream>>>(w2b, w2bT, DFFc,Dc);

    // -------- routing + pre-norm --------
    route_argmax_kernel<<<dim3((Tc + 255) / 256), blk, 0, stream>>>(ma, smp);
    layernorm_kernel<<<dim3(Tc), blk, 0, stream>>>(dec, g0, be0, x, xh);

    // q projection
    gemm_f16wmma_kernel<true,false,false,true,false>
        <<<dim3(Dc/128, Tc/128), blk, 0, stream>>>(xh, wqT, bq, nullptr, qb, nullptr, Tc, Dc, Dc);

    // K/V projections of the 64 unique memory entries (f32 A path)
    gemm_f16wmma_kernel<false,false,false,false,true>
        <<<dim3(Dc/128, MKc/128), blk, 0, stream>>>(enc, wkT, bk, nullptr, nullptr, kf16, MKc, Dc, Dc);
    gemm_f16wmma_kernel<false,false,false,false,true>
        <<<dim3(Dc/128, MKc/128), blk, 0, stream>>>(tgt, wvT, bv, nullptr, nullptr, vf16, MKc, Dc, Dc);

    // routed per-token attention
    routed_attn_kernel<<<dim3(Tc), blk, 0, stream>>>(qb, kf16, vf16, smp, maskmem, ctxh);

    // out projection + residual(x)
    gemm_f16wmma_kernel<true,false,true,true,true>
        <<<dim3(Dc/128, Tc/128), blk, 0, stream>>>(ctxh, woT, bo, x, st0, st0h, Tc, Dc, Dc);

    // FF-a
    gemm_f16wmma_kernel<true,true,false,false,true>
        <<<dim3(DFFc/128, Tc/128), blk, 0, stream>>>(st0h, w1aT, b1a, nullptr, nullptr, hh, Tc, DFFc, Dc);
    gemm_f16wmma_kernel<true,false,true,true,false>
        <<<dim3(Dc/128, Tc/128), blk, 0, stream>>>(hh, w2aT, b2a, st0, st1, nullptr, Tc, Dc, DFFc);

    // LN1 + gate + residual(x)
    ln_gate_kernel<<<dim3(Tc), blk, 0, stream>>>(st1, g1, be1, x, gl, yb, yh);

    // FF-b -> d_out
    gemm_f16wmma_kernel<true,true,false,false,true>
        <<<dim3(DFFc/128, Tc/128), blk, 0, stream>>>(yh, w1bT, b1b, nullptr, nullptr, hh, Tc, DFFc, Dc);
    gemm_f16wmma_kernel<true,false,true,true,false>
        <<<dim3(Dc/128, Tc/128), blk, 0, stream>>>(hh, w2bT, b2b, yb, (float*)d_out, nullptr, Tc, Dc, DFFc);
}
